// GraphSAGEModel_7816840478969
// MI455X (gfx1250) — compile-verified, matrix operand order checked by reading, not run on hardware
//
#include <hip/hip_runtime.h>
#include <hip/hip_bf16.h>

typedef __attribute__((ext_vector_type(16))) _Float16 v16h;
typedef __attribute__((ext_vector_type(8)))  float    v8f;

union Frag {
    v16h     h;
    unsigned u[8];
};

// ---------------------------------------------------------------------------
// zero a float region
// ---------------------------------------------------------------------------
__global__ void zero_f32(float* __restrict__ p, size_t count) {
    size_t i = (size_t)blockIdx.x * blockDim.x + threadIdx.x;
    if (i < count) p[i] = 0.0f;
}

// ---------------------------------------------------------------------------
// in-degree counts: cnt[dst[e]] += 1
// ---------------------------------------------------------------------------
__global__ void count_deg(const int* __restrict__ dst, float* __restrict__ cnt, int E) {
    int e = blockIdx.x * blockDim.x + threadIdx.x;
    if (e < E) atomicAdd(&cnt[dst[e]], 1.0f);
}

// ---------------------------------------------------------------------------
// scatter-add: agg[dst[e]][c] += feat[src[e]][c], c = 0..63
// one (edge, channel) pair per thread; both buffers are L2-resident (25.6 MB)
// ---------------------------------------------------------------------------
__global__ void scatter_feat(const float* __restrict__ feat,
                             const int* __restrict__ src,
                             const int* __restrict__ dst,
                             float* __restrict__ agg, int E) {
    long long t = (long long)blockIdx.x * blockDim.x + threadIdx.x;
    int e = (int)(t >> 6);
    int c = (int)(t & 63);
    if (e < E) {
        int s = src[e];
        int d = dst[e];
        atomicAdd(&agg[(size_t)d * 64 + c], feat[(size_t)s * 64 + c]);
    }
}

// ---------------------------------------------------------------------------
// SAGE layer GEMM, WMMA f16->f32:
//   h[n] = relu( (agg[n]/max(cnt[n],1)) @ Wl^T + bl + x[n] @ Wr^T )
// One wave per 16-node tile; 4 col-blocks x 4 WMMAs (2 K-steps x {Wl,Wr}).
// Weights staged in LDS as f16. Epilogue uses a wave-uniform full-tile test
// so the common path stores without per-lane exec masking.
// ---------------------------------------------------------------------------
__global__ void __launch_bounds__(128)
sage_gemm_relu(const float* __restrict__ xin, const float* __restrict__ agg,
               const float* __restrict__ cnt, const float* __restrict__ Wl,
               const float* __restrict__ bias, const float* __restrict__ Wr,
               float* __restrict__ hout, int n) {
    __shared__ _Float16 sWl[64 * 64];
    __shared__ _Float16 sWr[64 * 64];
    for (int i = threadIdx.x; i < 64 * 64; i += 128) {
        sWl[i] = (_Float16)Wl[i];
        sWr[i] = (_Float16)Wr[i];
    }
    __syncthreads();

    const int wave = threadIdx.x >> 5;
    const int lane = threadIdx.x & 31;
    const int tile = blockIdx.x * 4 + wave;
    if (tile * 16 >= n) return;                 // wave-uniform
    const bool full = (tile * 16 + 16) <= n;    // wave-uniform

    const int m    = lane & 15;   // row (A) / column (B) index within tile
    const int half = lane >> 4;

    int node = tile * 16 + m;
    if (node >= n) node = n - 1;  // clamp loads; ragged stores are guarded
    const float* xrow = xin + (size_t)node * 64;
    const float* arow = agg + (size_t)node * 64;
    const float  inv  = 1.0f / fmaxf(cnt[node], 1.0f);

    // K-pair base offsets for the 16-bit A/B fragment layout (pairs j=0..7):
    //   k(j) = (j/4)*16 + half*8 + (j%4)*2   (covers K=0..31 per fragment)
    int kp[8];
#pragma unroll
    for (int j = 0; j < 8; ++j)
        kp[j] = ((j >> 2) << 4) + (half << 3) + ((j & 3) << 1);

    // A fragments: mean rows (agg*inv) and raw x rows, K halves 0..31 / 32..63
    Frag aM[2], aX[2];
#pragma unroll
    for (int kb = 0; kb < 2; ++kb) {
#pragma unroll
        for (int j = 0; j < 8; ++j) {
            int k = kb * 32 + kp[j];
            aM[kb].h[2 * j]     = (_Float16)(arow[k] * inv);
            aM[kb].h[2 * j + 1] = (_Float16)(arow[k + 1] * inv);
            aX[kb].h[2 * j]     = (_Float16)xrow[k];
            aX[kb].h[2 * j + 1] = (_Float16)xrow[k + 1];
        }
    }

    const unsigned* wlU = (const unsigned*)sWl;
    const unsigned* wrU = (const unsigned*)sWr;

#pragma unroll
    for (int cb = 0; cb < 4; ++cb) {
        const int ncol = cb * 16 + m;          // output channel for this lane
        // B fragments: B[k][ncol] = W[ncol][k] -> contiguous W rows, f16 pairs
        Frag fl0, fl1, fr0, fr1;
#pragma unroll
        for (int j = 0; j < 8; ++j) {
            int base = ncol * 64 + kp[j];
            fl0.u[j] = wlU[base >> 1];
            fl1.u[j] = wlU[(base + 32) >> 1];
            fr0.u[j] = wrU[base >> 1];
            fr1.u[j] = wrU[(base + 32) >> 1];
        }
        v8f acc = {};
        acc = __builtin_amdgcn_wmma_f32_16x16x32_f16(false, aM[0].h, false, fl0.h,
                                                     (short)0, acc, false, false);
        acc = __builtin_amdgcn_wmma_f32_16x16x32_f16(false, aM[1].h, false, fl1.h,
                                                     (short)0, acc, false, false);
        acc = __builtin_amdgcn_wmma_f32_16x16x32_f16(false, aX[0].h, false, fr0.h,
                                                     (short)0, acc, false, false);
        acc = __builtin_amdgcn_wmma_f32_16x16x32_f16(false, aX[1].h, false, fr1.h,
                                                     (short)0, acc, false, false);

        const float b = bias[ncol];
        // C/D layout: VGPR r holds row M = r + 8*half, col N = lane&15
        float* outp = hout + (size_t)(tile * 16 + half * 8) * 64 + ncol;
        if (full) {
#pragma unroll
            for (int r = 0; r < 8; ++r)
                outp[(size_t)r * 64] = fmaxf(acc[r] + b, 0.0f);
        } else {
#pragma unroll
            for (int r = 0; r < 8; ++r) {
                int row = tile * 16 + r + half * 8;
                if (row < n)
                    hout[(size_t)row * 64 + ncol] = fmaxf(acc[r] + b, 0.0f);
            }
        }
    }
}

// ---------------------------------------------------------------------------
// per-node edge-scorer factorization:
//   sl[n] = h[n] . Wlin[0:64],  sr[n] = h[n] . Wlin[64:128]
// one wave per node, shuffle reduction
// ---------------------------------------------------------------------------
__global__ void __launch_bounds__(128)
node_score(const float* __restrict__ h, const float* __restrict__ Wlin,
           float* __restrict__ sl, float* __restrict__ sr, int n) {
    int wave = threadIdx.x >> 5;
    int lane = threadIdx.x & 31;
    int node = blockIdx.x * 4 + wave;
    if (node >= n) return;
    const float* row = h + (size_t)node * 64;
    float a0 = row[lane], a1 = row[lane + 32];
    float pl = a0 * Wlin[lane]      + a1 * Wlin[lane + 32];
    float pr = a0 * Wlin[64 + lane] + a1 * Wlin[96 + lane];
#pragma unroll
    for (int off = 16; off > 0; off >>= 1) {
        pl += __shfl_down(pl, off, 32);
        pr += __shfl_down(pr, off, 32);
    }
    if (lane == 0) {
        sl[node] = pl;
        sr[node] = pr;
    }
}

// ---------------------------------------------------------------------------
// out[e] = sl[src[e]] + sr[dst[e]] + blin
// ---------------------------------------------------------------------------
__global__ void edge_score(const int* __restrict__ src, const int* __restrict__ dst,
                           const float* __restrict__ sl, const float* __restrict__ sr,
                           const float* __restrict__ blin, float* __restrict__ out, int E) {
    int e = blockIdx.x * blockDim.x + threadIdx.x;
    if (e < E) out[e] = sl[src[e]] + sr[dst[e]] + blin[0];
}

// ---------------------------------------------------------------------------
extern "C" void kernel_launch(void* const* d_in, const int* in_sizes, int n_in,
                              void* d_out, int out_size, void* d_ws, size_t ws_size,
                              hipStream_t stream) {
    (void)n_in; (void)out_size; (void)ws_size;
    const float* x    = (const float*)d_in[0];
    const int*   ei   = (const int*)d_in[1];
    const float* W1l  = (const float*)d_in[2];
    const float* b1l  = (const float*)d_in[3];
    const float* W1r  = (const float*)d_in[4];
    const float* W2l  = (const float*)d_in[5];
    const float* b2l  = (const float*)d_in[6];
    const float* W2r  = (const float*)d_in[7];
    const float* Wlin = (const float*)d_in[8];
    const float* blin = (const float*)d_in[9];

    const int n = in_sizes[0] / 64;   // nodes
    const int E = in_sizes[1] / 2;    // edges
    const int* src = ei;
    const int* dst = ei + E;

    // workspace layout (floats): agg[64n] | cnt[n] | h1[64n] | h2[64n] | sl[n] | sr[n]
    float* agg = (float*)d_ws;
    float* cnt = agg + (size_t)n * 64;
    float* h1  = cnt + n;
    float* h2  = h1 + (size_t)n * 64;
    float* sl  = h2 + (size_t)n * 64;
    float* sr  = sl + n;
    float* out = (float*)d_out;

    const int tiles       = (n + 15) / 16;
    const int gemm_blocks = (tiles + 3) / 4;
    const long long sc_t  = (long long)E * 64;
    const int sc_blocks   = (int)((sc_t + 255) / 256);
    const int eb          = (E + 255) / 256;

    // ---- conv1 ----
    {
        size_t z = (size_t)n * 65;  // agg + cnt (contiguous)
        zero_f32<<<(unsigned)((z + 255) / 256), 256, 0, stream>>>(agg, z);
    }
    count_deg<<<eb, 256, 0, stream>>>(dst, cnt, E);
    scatter_feat<<<sc_blocks, 256, 0, stream>>>(x, src, dst, agg, E);
    sage_gemm_relu<<<gemm_blocks, 128, 0, stream>>>(x, agg, cnt, W1l, b1l, W1r, h1, n);

    // ---- conv2 (reuse cnt; re-zero agg only) ----
    {
        size_t z = (size_t)n * 64;
        zero_f32<<<(unsigned)((z + 255) / 256), 256, 0, stream>>>(agg, z);
    }
    scatter_feat<<<sc_blocks, 256, 0, stream>>>(h1, src, dst, agg, E);
    sage_gemm_relu<<<gemm_blocks, 128, 0, stream>>>(h1, agg, cnt, W2l, b2l, W2r, h2, n);

    // ---- edge scorer ----
    node_score<<<(n + 3) / 4, 128, 0, stream>>>(h2, Wlin, sl, sr, n);
    edge_score<<<eb, 256, 0, stream>>>(src, dst, sl, sr, blin, out, E);
}
